// RingDilatedAttentionV3_70420283785271
// MI455X (gfx1250) — matrix operand order repaired and checked by reading, not compiled, not run on hardware
//
#include <hip/hip_runtime.h>

typedef __attribute__((ext_vector_type(16))) _Float16 v16h;
typedef __attribute__((ext_vector_type(8)))  _Float16 v8h;
typedef __attribute__((ext_vector_type(2)))  _Float16 v2h;
typedef __attribute__((ext_vector_type(8)))  float    v8f;
typedef __attribute__((ext_vector_type(8)))  int      v8i;

#define N_TOK   8192
#define N_HEAD  12
#define DIM     64
#define ROWSTR  (N_HEAD * DIM)   /* 768 floats per token */
#define SK_STR  72               /* halves per key row in sK  (144B, 16B aligned) */
#define SVT_STR 72               /* halves per d  row in sVt  (144B, 16B aligned) */
#define QSCALE  0.18033688011112042f  /* (1/sqrt(64)) * log2(e): log2-domain scores */

// cross-half (lane ^ 16) swap as pure VALU (v_permlanex16), no LDS pipe.
__device__ __forceinline__ int xhalf_i(int x) {
  return __builtin_amdgcn_permlanex16(x, x, 0x76543210, 0xfedcba98, false, false);
}
__device__ __forceinline__ float xhalf_f(float x) {
  return __builtin_bit_cast(float, xhalf_i(__builtin_bit_cast(int, x)));
}

// Effective attention per (b, head, segment): s queries x 512 distinct dilated
// keys (dilation repeats cancel exactly in softmax). One wave = 16 queries,
// flash-style over 8 chunks of 64 keys, matmuls via v_wmma_f32_16x16x32_f16.
__global__ __launch_bounds__(256)
void dilated_attn_wmma(const float* __restrict__ Q,
                       const float* __restrict__ K,
                       const float* __restrict__ V,
                       float* __restrict__ Out) {
  __shared__ __align__(16) _Float16 sK [64 * SK_STR];   // K chunk, [key][d], f16
  __shared__ __align__(16) _Float16 sVt[64 * SVT_STR];  // V chunk transposed, [d][key], f16

  const int tid  = threadIdx.x;
  const int lane = tid & 31;
  const int wv   = tid >> 5;     // wave 0..7
  const int col  = lane & 15;    // tile column (query) / tile row index
  const int lh   = lane >> 4;    // lane half (0/1)

  const int wg  = blockIdx.x;                // b*768 + h*64 + t64
  const int b   = wg / (N_HEAD * 64);
  const int rem = wg - b * (N_HEAD * 64);
  const int h   = rem >> 6;                  // head 0..11
  const int t64 = rem & 63;
  const int cfg = h >> 2;                    // 0,1,2
  const int s   = 512 << cfg;                // segment length
  const int r   = 1 << cfg;                  // dilation rate
  const int qbs = 4 << cfg;                  // 128-query blocks per segment
  const int seg = t64 / qbs;
  const int qb  = t64 - seg * qbs;
  const int segbase = seg * s;
  const int off = (h & 3) & (r - 1);         // per-head dilation offset

  const size_t bofs = (size_t)b * N_TOK * ROWSTR;
  const int qrow = segbase + qb * 128 + wv * 16 + col;

  // ---- Q -> B-operand registers (Q^T), pre-scaled, f32->f16 ----
  // B layout (32x16): lane: N=col, element e -> k = lh*16 + e (16 consecutive d)
  const float* qp = Q + bofs + (size_t)qrow * ROWSTR + h * DIM;
  v16h Qb[2];
  #pragma unroll
  for (int dc = 0; dc < 2; ++dc) {
    #pragma unroll
    for (int e = 0; e < 16; ++e)
      Qb[dc][e] = (_Float16)(qp[dc * 32 + lh * 16 + e] * QSCALE);
  }

  v8f O[4] = {};                 // out^T tiles: d = dt*16 + j + 8*lh, q = col
  float m_run = -1e30f;
  float l_run = 0.0f;

  const float* kb = K + bofs + h * DIM;
  const float* vb = V + bofs + h * DIM;

  for (int ch = 0; ch < 8; ++ch) {
    __syncthreads();
    // ---- stage K chunk: [key][d] f16 (each thread: 1 key row, 16 d) ----
    {
      const int key = tid >> 2;              // 0..63
      const int db  = (tid & 3) << 4;        // 0,16,32,48
      const int row = segbase + off + (ch * 64 + key) * r;
      const float* kp = kb + (size_t)row * ROWSTR + db;
      v8h k0, k1;
      #pragma unroll
      for (int i = 0; i < 8; ++i) { k0[i] = (_Float16)kp[i]; k1[i] = (_Float16)kp[8 + i]; }
      *(v8h*)&sK[key * SK_STR + db]     = k0;
      *(v8h*)&sK[key * SK_STR + db + 8] = k1;
    }
    // ---- stage V chunk transposed: [d][key] f16 (each thread: 1 d, 16 keys) ----
    {
      const int d  = tid & 63;
      const int kq = tid >> 6;               // 0..3 -> keys kq*16..kq*16+15
      const float* vp = vb + d;
      v8h t0, t1;
      #pragma unroll
      for (int i = 0; i < 8; ++i) {
        const int key = kq * 16 + i;
        t0[i] = (_Float16)vp[(size_t)(segbase + off + (ch * 64 + key) * r) * ROWSTR];
        t1[i] = (_Float16)vp[(size_t)(segbase + off + (ch * 64 + key + 8) * r) * ROWSTR];
      }
      *(v8h*)&sVt[d * SVT_STR + kq * 16]     = t0;
      *(v8h*)&sVt[d * SVT_STR + kq * 16 + 8] = t1;
    }
    __syncthreads();

    // ---- S^T = K_chunk · Q^T : four 16(key)x16(query) tiles (log2 domain) ----
    v8f St[4];
    #pragma unroll
    for (int c = 0; c < 4; ++c) {
      St[c] = v8f{};
      const int key = c * 16 + col;
      #pragma unroll
      for (int dc = 0; dc < 2; ++dc) {
        // A layout (16x32): lane: M=key, e -> d32 = lh*16+e
        const _Float16* pa = &sK[key * SK_STR + dc * 32 + lh * 16];
        v8h lo = *(const v8h*)pa;
        v8h hi = *(const v8h*)(pa + 8);
        v16h A;
        #pragma unroll
        for (int e = 0; e < 8; ++e) { A[e] = lo[e]; A[e + 8] = hi[e]; }
        St[c] = __builtin_amdgcn_wmma_f32_16x16x32_f16(
            false, A, false, Qb[dc], (short)0, St[c], false, false);
      }
    }

    // ---- online softmax, base-2 (stats per query == per lane) ----
    float cm = -1e30f;
    #pragma unroll
    for (int c = 0; c < 4; ++c)
      #pragma unroll
      for (int j = 0; j < 8; ++j) cm = fmaxf(cm, St[c][j]);
    cm = fmaxf(cm, xhalf_f(cm));
    const float nm    = fmaxf(m_run, cm);
    const float alpha = __builtin_amdgcn_exp2f(m_run - nm);
    m_run = nm;

    float p[4][8];
    float rs = 0.0f;
    #pragma unroll
    for (int c = 0; c < 4; ++c)
      #pragma unroll
      for (int j = 0; j < 8; ++j) {
        p[c][j] = __builtin_amdgcn_exp2f(St[c][j] - nm);
        rs += p[c][j];
      }
    rs += xhalf_f(rs);
    l_run = l_run * alpha + rs;
    #pragma unroll
    for (int dt = 0; dt < 4; ++dt)
      #pragma unroll
      for (int j = 0; j < 8; ++j) O[dt][j] *= alpha;

    // ---- build P^T B-operands (two: keys 0-31 from tiles 0,1; 32-63 from 2,3)
    // packed f16 pair shuffles: e<8: lh ? xhalf(hi-tile) : lo-tile
    //                           e>=8: lh ? hi-tile : xhalf(lo-tile)
    v16h Pb[2];
    #pragma unroll
    for (int kc = 0; kc < 2; ++kc) {
      int u0[4], u1[4];
      #pragma unroll
      for (int i = 0; i < 4; ++i) {
        v2h a, bq;
        a[0]  = (_Float16)p[2 * kc][2 * i];     a[1]  = (_Float16)p[2 * kc][2 * i + 1];
        bq[0] = (_Float16)p[2 * kc + 1][2 * i]; bq[1] = (_Float16)p[2 * kc + 1][2 * i + 1];
        u0[i] = __builtin_bit_cast(int, a);
        u1[i] = __builtin_bit_cast(int, bq);
      }
      v8i pw;
      #pragma unroll
      for (int i = 0; i < 4; ++i) {
        const int x1 = xhalf_i(u1[i]);
        const int x0 = xhalf_i(u0[i]);
        pw[i]     = lh ? x1    : u0[i];
        pw[i + 4] = lh ? u1[i] : x0;
      }
      Pb[kc] = __builtin_bit_cast(v16h, pw);
    }

    // ---- out^T += V^T · P^T ----
    #pragma unroll
    for (int dt = 0; dt < 4; ++dt) {
      #pragma unroll
      for (int kc = 0; kc < 2; ++kc) {
        // A layout: lane: M = d = dt*16+col, e -> key = kc*32+lh*16+e
        const _Float16* pa = &sVt[(dt * 16 + col) * SVT_STR + kc * 32 + lh * 16];
        v8h lo = *(const v8h*)pa;
        v8h hi = *(const v8h*)(pa + 8);
        v16h A;
        #pragma unroll
        for (int e = 0; e < 8; ++e) { A[e] = lo[e]; A[e + 8] = hi[e]; }
        O[dt] = __builtin_amdgcn_wmma_f32_16x16x32_f16(
            false, A, false, Pb[kc], (short)0, O[dt], false, false);
      }
    }
  }

  // ---- normalize and store: lane holds 8 consecutive d per d-tile ----
  const float rcp = __builtin_amdgcn_rcpf(l_run);
  float* op = Out + bofs + (size_t)qrow * ROWSTR + h * DIM;
  #pragma unroll
  for (int dt = 0; dt < 4; ++dt) {
    const int d0 = dt * 16 + lh * 8;
    #pragma unroll
    for (int j = 0; j < 8; ++j) op[d0 + j] = O[dt][j] * rcp;
  }
}

extern "C" void kernel_launch(void* const* d_in, const int* in_sizes, int n_in,
                              void* d_out, int out_size, void* d_ws, size_t ws_size,
                              hipStream_t stream) {
  const float* q = (const float*)d_in[0];
  const float* k = (const float*)d_in[1];
  const float* v = (const float*)d_in[2];
  float* out = (float*)d_out;
  const int B = in_sizes[0] / (N_TOK * ROWSTR);   // = 2
  dim3 grid(B * N_HEAD * 64);
  dim3 block(256);
  hipLaunchKernelGGL(dilated_attn_wmma, grid, block, 0, stream, q, k, v, out);
}
